// DGCNN_72172630442645
// MI455X (gfx1250) — matrix operand order repaired and checked by reading, not compiled, not run on hardware
//
#include <hip/hip_runtime.h>

#define B_   8
#define P_   4096
#define KNN  20
#define H_   128
#define INF_ 3.0e38f

typedef __attribute__((ext_vector_type(16))) _Float16 v16h;
typedef __attribute__((ext_vector_type(8)))  float    v8f;

// 16-bit A-matrix 16x32 element->K mapping (per ISA 7.12.2):
// lane group g = lane>>4; vector element e in [0,16)
// g==0: e<8 -> K=e,    e>=8 -> K=e+8
// g==1: e<8 -> K=e+8,  e>=8 -> K=e+16
__device__ __forceinline__ int a_kmap(int e, int g) {
    return (e & 7) + 8 * g + 16 * (e >> 3);
}

// ---------------------------------------------------------------------------
// Layer 1: brute-force kNN in 3-D input space (D=3, scalar path)
// ---------------------------------------------------------------------------
__global__ void knn3_kernel(const float* __restrict__ x, int* __restrict__ idx) {
    int t = blockIdx.x * blockDim.x + threadIdx.x;
    if (t >= B_ * P_) return;
    int b = t / P_, i = t % P_;
    const float* xb = x + (size_t)b * P_ * 3;
    float x0 = xb[i * 3], x1 = xb[i * 3 + 1], x2 = xb[i * 3 + 2];
    float bd[KNN]; int bi[KNN];
    for (int k = 0; k < KNN; ++k) { bd[k] = INF_; bi[k] = 0; }
    for (int j = 0; j < P_; ++j) {
        if (j == i) continue;
        float d0 = xb[j * 3]     - x0;
        float d1 = xb[j * 3 + 1] - x1;
        float d2v = xb[j * 3 + 2] - x2;
        float d = d0 * d0 + d1 * d1 + d2v * d2v;
        if (d < bd[KNN - 1]) {
            int pos = KNN - 1;
            while (pos > 0 && bd[pos - 1] > d) {
                bd[pos] = bd[pos - 1]; bi[pos] = bi[pos - 1]; --pos;
            }
            bd[pos] = d; bi[pos] = j;
        }
    }
    int* o = idx + (size_t)t * KNN;
    for (int k = 0; k < KNN; ++k) o[k] = bi[k];
}

// ---------------------------------------------------------------------------
// Layer 1 edge MLP (2D=6 inputs -> 128 outputs): scalar, 128 threads / point
// ---------------------------------------------------------------------------
__global__ void mlp1_kernel(const float* __restrict__ x, const int* __restrict__ idx,
                            const float* __restrict__ W1, const float* __restrict__ b1,
                            float* __restrict__ out_f32, _Float16* __restrict__ out_f16) {
    int p = blockIdx.x;      // 0 .. B*P-1
    int t = threadIdx.x;     // 0 .. 127 (output channel)
    int b = p / P_, i = p % P_;
    const float* xb = x + (size_t)b * P_ * 3;
    float xi0 = xb[i * 3], xi1 = xb[i * 3 + 1], xi2 = xb[i * 3 + 2];
    float w0 = W1[0 * H_ + t], w1 = W1[1 * H_ + t], w2 = W1[2 * H_ + t];
    float w3 = W1[3 * H_ + t], w4 = W1[4 * H_ + t], w5 = W1[5 * H_ + t];
    float bb = b1[t];
    const int* nb = idx + (size_t)p * KNN;
    float m = -INF_;
    for (int k = 0; k < KNN; ++k) {
        int j = nb[k];
        float d0 = xb[j * 3]     - xi0;
        float d1 = xb[j * 3 + 1] - xi1;
        float d2v = xb[j * 3 + 2] - xi2;
        float v = xi0 * w0 + xi1 * w1 + xi2 * w2 + d0 * w3 + d1 * w4 + d2v * w5 + bb;
        v = v > 0.f ? v : 0.f;
        m = fmaxf(m, v);
    }
    out_f32[(size_t)p * H_ + t] = m;
    out_f16[(size_t)p * H_ + t] = (_Float16)m;
}

// ---------------------------------------------------------------------------
// Utilities
// ---------------------------------------------------------------------------
__global__ void sqnorm_kernel(const float* __restrict__ feat, float* __restrict__ sq) {
    int t = blockIdx.x * blockDim.x + threadIdx.x;
    if (t >= B_ * P_) return;
    const float* f = feat + (size_t)t * H_;
    float s = 0.f;
    for (int c = 0; c < H_; ++c) s += f[c] * f[c];
    sq[t] = s;
}

__global__ void cvt_f16_kernel(const float* __restrict__ src, _Float16* __restrict__ dst, int n) {
    int t = blockIdx.x * blockDim.x + threadIdx.x;
    if (t < n) dst[t] = (_Float16)src[t];
}

// ---------------------------------------------------------------------------
// Distance tile via WMMA: d2[i][j] = |xi|^2 + |xj|^2 - 2 * (X X^T)[i][j]
// One wave per 16x16 tile; K=128 in 4 steps of v_wmma_f32_16x16x32_f16.
// 64MB d2 buffer per batch stays resident in the 192MB L2.
// ---------------------------------------------------------------------------
__global__ void gram_d2_kernel(const _Float16* __restrict__ feat16,
                               const float* __restrict__ sq,
                               float* __restrict__ d2, int batch) {
    const int lane = threadIdx.x;
    const int i0 = blockIdx.x * 16;
    const int j0 = blockIdx.y * 16;
    const _Float16* Xb = feat16 + (size_t)batch * P_ * H_;
    const float* sqb = sq + (size_t)batch * P_;
    const int ncol = lane & 15, grp = lane >> 4;
    v8f acc = {};
    for (int ks = 0; ks < 4; ++ks) {
        int k0 = ks * 32;
        v16h a, bt;
#pragma unroll
        for (int e = 0; e < 16; ++e) {
            a[e]  = Xb[(size_t)(i0 + ncol) * H_ + k0 + a_kmap(e, grp)];
            bt[e] = Xb[(size_t)(j0 + ncol) * H_ + k0 + a_kmap(e, grp)];  // B[k][n] = X[j0+n][k]
        }
        acc = __builtin_amdgcn_wmma_f32_16x16x32_f16(false, a, false, bt,
                                                     (short)0, acc, false, false);
    }
#pragma unroll
    for (int r = 0; r < 8; ++r) {
        int row = i0 + r + 8 * grp;   // C layout: M = vgpr + 8*(lane/16)
        int col = j0 + ncol;          //           N = lane%16
        float v = sqb[row] + sqb[col] - 2.f * acc[r];
        if (row == col) v = INF_;     // exclude self-loops
        d2[(size_t)row * P_ + col] = v;
    }
}

// ---------------------------------------------------------------------------
// Top-K (K=20 smallest) per row of the per-batch d2 matrix (reads hit L2)
// ---------------------------------------------------------------------------
__global__ void topk_kernel(const float* __restrict__ d2, int* __restrict__ idx, int batch) {
    int i = blockIdx.x * blockDim.x + threadIdx.x;
    if (i >= P_) return;
    float bd[KNN]; int bi[KNN];
    for (int k = 0; k < KNN; ++k) { bd[k] = INF_; bi[k] = 0; }
    const float* row = d2 + (size_t)i * P_;
    for (int j = 0; j < P_; ++j) {
        float d = row[j];
        if (d < bd[KNN - 1]) {
            int pos = KNN - 1;
            while (pos > 0 && bd[pos - 1] > d) {
                bd[pos] = bd[pos - 1]; bi[pos] = bi[pos - 1]; --pos;
            }
            bd[pos] = d; bi[pos] = j;
        }
    }
    int* o = idx + ((size_t)batch * P_ + i) * KNN;
    for (int k = 0; k < KNN; ++k) o[k] = bi[k];
}

// ---------------------------------------------------------------------------
// EdgeConv MLP via WMMA: per point, E[20->32 pad][256] @ W[256][128],
// bias + ReLU, masked max over the 20 valid edge rows.
// One wave per point; edge matrix staged in 16KB LDS.
// ---------------------------------------------------------------------------
__global__ void edge_mlp_wmma_kernel(const _Float16* __restrict__ feat16,
                                     const int* __restrict__ idx,
                                     const _Float16* __restrict__ Wf16,
                                     const float* __restrict__ bias,
                                     float* __restrict__ out_f32,
                                     _Float16* __restrict__ out_f16) {
    __shared__ _Float16 E[32][2 * H_];           // 32 x 256 f16 = 16 KB
    const int p = blockIdx.x;                    // 0 .. B*P-1
    const int lane = threadIdx.x;                // 0 .. 31
    const int b = p / P_, i = p % P_;
    const _Float16* xi = feat16 + (size_t)p * H_;
    const int* nb = idx + ((size_t)b * P_ + i) * KNN;

    for (int m = 0; m < 32; ++m) {
        if (m < KNN) {
            const _Float16* xj = feat16 + ((size_t)b * P_ + nb[m]) * H_;
#pragma unroll
            for (int cc = 0; cc < 4; ++cc) {
                int c = lane + cc * 32;
                E[m][c]      = xi[c];                       // [xi, ...]
                E[m][c + H_] = (_Float16)(xj[c] - xi[c]);   // [..., xj-xi]
            }
        } else {
#pragma unroll
            for (int cc = 0; cc < 4; ++cc) {
                int c = lane + cc * 32;
                E[m][c] = (_Float16)0.f;
                E[m][c + H_] = (_Float16)0.f;
            }
        }
    }
    __syncthreads();

    const int ncol = lane & 15;
    const int grp  = lane >> 4;
    for (int nt = 0; nt < 8; ++nt) {             // 8 N-tiles of 16 (H=128)
        v8f acc0 = {}, acc1 = {};
        for (int ks = 0; ks < 8; ++ks) {         // K = 256 in steps of 32
            const int k0 = ks * 32;
            v16h a0, a1, bt;
#pragma unroll
            for (int e = 0; e < 16; ++e) {
                int ka = k0 + a_kmap(e, grp);
                a0[e] = E[ncol][ka];             // M-tile 0: edges 0..15
                a1[e] = E[16 + ncol][ka];        // M-tile 1: edges 16..31
                int kb = k0 + a_kmap(e, grp);
                bt[e] = Wf16[(size_t)kb * H_ + nt * 16 + ncol];
            }
            acc0 = __builtin_amdgcn_wmma_f32_16x16x32_f16(false, a0, false, bt,
                                                          (short)0, acc0, false, false);
            acc1 = __builtin_amdgcn_wmma_f32_16x16x32_f16(false, a1, false, bt,
                                                          (short)0, acc1, false, false);
        }
        float bn = bias[nt * 16 + ncol];
        float m = -INF_;
#pragma unroll
        for (int r = 0; r < 8; ++r) {
            int row0 = r + 8 * grp;              // edge index of acc0[r]
            float v0 = acc0[r] + bn; v0 = v0 > 0.f ? v0 : 0.f;
            if (row0 < KNN) m = fmaxf(m, v0);
            int row1 = 16 + row0;                // edge index of acc1[r]
            float v1 = acc1[r] + bn; v1 = v1 > 0.f ? v1 : 0.f;
            if (row1 < KNN) m = fmaxf(m, v1);
        }
        // columns are split between lane l and lane l+16 -> combine
        m = fmaxf(m, __shfl_xor(m, 16, 32));
        if (lane < 16) {
            out_f32[(size_t)p * H_ + nt * 16 + lane] = m;
            out_f16[(size_t)p * H_ + nt * 16 + lane] = (_Float16)m;
        }
    }
}

// ---------------------------------------------------------------------------
// Global mean pool + dense head (tiny)
// ---------------------------------------------------------------------------
__global__ void meanpool_kernel(const float* __restrict__ feat, float* __restrict__ pooled) {
    int t = blockIdx.x * blockDim.x + threadIdx.x;   // B*H
    if (t >= B_ * H_) return;
    int b = t / H_, c = t % H_;
    float s = 0.f;
    for (int p = 0; p < P_; ++p) s += feat[((size_t)b * P_ + p) * H_ + c];
    pooled[t] = s * (1.0f / (float)P_);
}

__global__ void head_kernel(const float* __restrict__ pooled,
                            const float* __restrict__ W4, const float* __restrict__ b4,
                            const float* __restrict__ W5, const float* __restrict__ b5,
                            float* __restrict__ out) {
    __shared__ float g[B_][H_];
    int t = threadIdx.x;                             // 128 threads
    for (int b = 0; b < B_; ++b) {
        float s = b4[t];
        for (int d = 0; d < H_; ++d) s += pooled[b * H_ + d] * W4[d * H_ + t];
        g[b][t] = s > 0.f ? s : 0.f;
    }
    __syncthreads();
    if (t < B_ * 3) {
        int b = t / 3, o = t % 3;
        float s = b5[o];
        for (int c = 0; c < H_; ++c) s += g[b][c] * W5[c * 3 + o];
        out[b * 3 + o] = s;
    }
}

// ---------------------------------------------------------------------------
// Orchestration
// ---------------------------------------------------------------------------
extern "C" void kernel_launch(void* const* d_in, const int* in_sizes, int n_in,
                              void* d_out, int out_size, void* d_ws, size_t ws_size,
                              hipStream_t stream) {
    const float* x  = (const float*)d_in[0];
    const float* W1 = (const float*)d_in[1];
    const float* b1 = (const float*)d_in[2];
    const float* W2 = (const float*)d_in[3];
    const float* b2 = (const float*)d_in[4];
    const float* W3 = (const float*)d_in[5];
    const float* b3 = (const float*)d_in[6];
    const float* W4 = (const float*)d_in[7];
    const float* b4 = (const float*)d_in[8];
    const float* W5 = (const float*)d_in[9];
    const float* b5 = (const float*)d_in[10];
    float* out = (float*)d_out;
    (void)in_sizes; (void)n_in; (void)out_size; (void)ws_size;

    char* ws = (char*)d_ws;
    auto alloc = [&](size_t bytes) {
        void* ptr = (void*)ws;
        ws += (bytes + 255) & ~(size_t)255;
        return ptr;
    };
    float*    featA   = (float*)   alloc(sizeof(float)    * (size_t)B_ * P_ * H_);  // 16 MB
    float*    featB   = (float*)   alloc(sizeof(float)    * (size_t)B_ * P_ * H_);  // 16 MB
    _Float16* featA16 = (_Float16*)alloc(sizeof(_Float16) * (size_t)B_ * P_ * H_);  //  8 MB
    _Float16* featB16 = (_Float16*)alloc(sizeof(_Float16) * (size_t)B_ * P_ * H_);  //  8 MB
    float*    sq      = (float*)   alloc(sizeof(float)    * (size_t)B_ * P_);
    float*    d2buf   = (float*)   alloc(sizeof(float)    * (size_t)P_ * P_);       // 64 MB (fits L2)
    int*      idx     = (int*)     alloc(sizeof(int)      * (size_t)B_ * P_ * KNN);
    _Float16* wf16    = (_Float16*)alloc(sizeof(_Float16) * 2 * H_ * H_);
    float*    pooled  = (float*)   alloc(sizeof(float)    * B_ * H_);

    // ---- Layer 1 (D=3): scalar kNN + scalar edge MLP ----
    knn3_kernel<<<(B_ * P_ + 127) / 128, 128, 0, stream>>>(x, idx);
    mlp1_kernel<<<B_ * P_, H_, 0, stream>>>(x, idx, W1, b1, featA, featA16);

    // ---- Layers 2 & 3 (D=128): WMMA Gram + top-K + WMMA edge MLP ----
    const float* Ws[2] = { W2, W3 };
    const float* bs[2] = { b2, b3 };
    float* cur = featA; _Float16* cur16 = featA16;
    float* nxt = featB; _Float16* nxt16 = featB16;
    for (int layer = 0; layer < 2; ++layer) {
        cvt_f16_kernel<<<(2 * H_ * H_ + 255) / 256, 256, 0, stream>>>(Ws[layer], wf16, 2 * H_ * H_);
        sqnorm_kernel<<<(B_ * P_ + 127) / 128, 128, 0, stream>>>(cur, sq);
        for (int b = 0; b < B_; ++b) {   // per-batch so the 64MB d2 stays in L2
            gram_d2_kernel<<<dim3(P_ / 16, P_ / 16), 32, 0, stream>>>(cur16, sq, d2buf, b);
            topk_kernel<<<(P_ + 127) / 128, 128, 0, stream>>>(d2buf, idx, b);
        }
        edge_mlp_wmma_kernel<<<B_ * P_, 32, 0, stream>>>(cur16, idx, wf16, bs[layer], nxt, nxt16);
        float* tf = cur; cur = nxt; nxt = tf;
        _Float16* th = cur16; cur16 = nxt16; nxt16 = th;
    }

    // ---- Pool + head ----
    meanpool_kernel<<<(B_ * H_ + 127) / 128, 128, 0, stream>>>(cur, pooled);
    head_kernel<<<1, H_, 0, stream>>>(pooled, W4, b4, W5, b5, out);
}
